// BlockAttentionResidual_40432822125108
// MI455X (gfx1250) — compile-verified, matrix operand order checked by reading, not run on hardware
//
#include <hip/hip_runtime.h>
#include <hip/hip_bf16.h>
#include <stdint.h>


typedef __attribute__((ext_vector_type(8)))  int      v8i;
typedef __attribute__((ext_vector_type(8)))  float    v8f;
typedef __attribute__((ext_vector_type(8)))  _Float16 v8h;
typedef __attribute__((ext_vector_type(16))) _Float16 v16h;

#define B_   2
#define S_   4096
#define H_   2048
#define NH_  16
#define HD_  128
#define NB_  8
#define BT_  512
#define INTER_ 4096
#define M_   (B_*S_)
#define QKV_N 6144
#define UP_N  8192

__device__ __forceinline__ v16h cat8(v8h lo, v8h hi) {
  return __builtin_shufflevector(lo, hi, 0,1,2,3,4,5,6,7,8,9,10,11,12,13,14,15);
}

__device__ __forceinline__ void async_copy32(unsigned lds, unsigned long long g) {
  asm volatile("global_load_async_to_lds_b128 %0, %1, off\n\t"
               "global_load_async_to_lds_b128 %0, %1, off offset:16"
               :: "v"(lds), "v"(g) : "memory");
}
__device__ __forceinline__ void async_wait0() {
  asm volatile("s_wait_asynccnt 0x0" ::: "memory");
}
__device__ __forceinline__ unsigned lds_off(const void* p) {
  return (unsigned)(uintptr_t)p;
}

template<int OP>
__device__ __forceinline__ float blockReduce(float v) {
  __shared__ float red[8];
  const int lane = threadIdx.x & 31, wid = threadIdx.x >> 5;
#pragma unroll
  for (int m = 16; m >= 1; m >>= 1) {
    float o = __shfl_xor(v, m, 32);
    v = OP ? fmaxf(v, o) : (v + o);
  }
  __syncthreads();
  if (lane == 0) red[wid] = v;
  __syncthreads();
  v = red[0];
#pragma unroll
  for (int i = 1; i < 8; ++i) v = OP ? fmaxf(v, red[i]) : (v + red[i]);
  return v;
}

__global__ __launch_bounds__(256) void absum_part_kernel(
    const float* __restrict__ W, float* __restrict__ part, int n) {
  const long long b0 = (long long)blockIdx.x * 65536;
  float s = 0.f;
  for (int k = 0; k < 256; ++k) {
    long long i = b0 + k * 256 + threadIdx.x;
    if (i < n) s += fabsf(W[i]);
  }
  s = blockReduce<0>(s);
  if (threadIdx.x == 0) part[blockIdx.x] = s;
}

__global__ __launch_bounds__(256) void absum_final_kernel(
    const float* __restrict__ part, int nparts, float* __restrict__ scalep, float count) {
  float s = 0.f;
  for (int i = threadIdx.x; i < nparts; i += 256) s += part[i];
  s = blockReduce<0>(s);
  if (threadIdx.x == 0) {
    float mean = s / count;
    *scalep = 1.f / fmaxf(mean, 1e-5f);
  }
}

__global__ __launch_bounds__(256) void wquant_kernel(
    const float* __restrict__ W, const float* __restrict__ sp,
    signed char* __restrict__ WQ, int n) {
  const float s = *sp;
  for (int i = blockIdx.x * 256 + threadIdx.x; i < n; i += gridDim.x * 256) {
    float q = rintf(W[i] * s);
    q = fminf(1.f, fmaxf(-1.f, q));
    WQ[i] = (signed char)q;
  }
}

__global__ __launch_bounds__(256) void cvt16_kernel(
    const float* __restrict__ W, _Float16* __restrict__ O, int n) {
  for (int i = blockIdx.x * 256 + threadIdx.x; i < n; i += gridDim.x * 256)
    O[i] = (_Float16)W[i];
}

template<int MODE>
__global__ __launch_bounds__(256) void rowop_kernel(
    const float* __restrict__ X, const float* __restrict__ nw,
    signed char* __restrict__ Q, _Float16* __restrict__ O16,
    float* __restrict__ SX) {
  const int row = blockIdx.x, tid = threadIdx.x;
  const float* xr = X + (size_t)row * H_;
  float v[8];
  float ss = 0.f;
#pragma unroll
  for (int j = 0; j < 8; ++j) {
    v[j] = xr[j * 256 + tid];
    if (MODE != 1) ss += v[j] * v[j];
  }
  if (MODE != 1) {
    ss = blockReduce<0>(ss);
    float r = __frsqrt_rn(ss * (1.0f / H_) + 1e-5f);
#pragma unroll
    for (int j = 0; j < 8; ++j) v[j] = v[j] * r * nw[j * 256 + tid];
  }
  if (MODE == 2) {
#pragma unroll
    for (int j = 0; j < 8; ++j) O16[(size_t)row * H_ + j * 256 + tid] = (_Float16)v[j];
    return;
  }
  float am = 0.f;
#pragma unroll
  for (int j = 0; j < 8; ++j) am = fmaxf(am, fabsf(v[j]));
  am = blockReduce<1>(am);
  const float s = 127.0f / fmaxf(am, 1e-5f);
  if (tid == 0) SX[row] = s;
#pragma unroll
  for (int j = 0; j < 8; ++j) {
    float q = rintf(v[j] * s);
    q = fminf(127.f, fmaxf(-128.f, q));
    Q[(size_t)row * H_ + j * 256 + tid] = (signed char)q;
  }
}

__global__ __launch_bounds__(256) void silu_quant_kernel(
    const _Float16* __restrict__ GV, signed char* __restrict__ TQ,
    float* __restrict__ ST) {
  const int row = blockIdx.x, tid = threadIdx.x;
  const _Float16* gr = GV + (size_t)row * (2 * INTER_);
  float t[16];
  float am = 0.f;
#pragma unroll
  for (int j = 0; j < 16; ++j) {
    const int col = j * 256 + tid;
    float g  = (float)gr[col];
    float vv = (float)gr[INTER_ + col];
    float si = g / (1.f + __expf(-g));
    t[j] = si * vv;
    am = fmaxf(am, fabsf(t[j]));
  }
  am = blockReduce<1>(am);
  const float s = 127.0f / fmaxf(am, 1e-5f);
  if (tid == 0) ST[row] = s;
#pragma unroll
  for (int j = 0; j < 16; ++j) {
    float q = rintf(t[j] * s);
    q = fminf(127.f, fmaxf(-128.f, q));
    TQ[(size_t)row * INTER_ + j * 256 + tid] = (signed char)q;
  }
}

template<bool OUTF16, bool RESID>
__global__ __launch_bounds__(256) void gemm_iu8_kernel(
    const signed char* __restrict__ A, const signed char* __restrict__ W,
    const float* __restrict__ sx, const float* __restrict__ swp,
    const float* __restrict__ resid, void* __restrict__ out,
    int M, int N, int K) {
  __shared__ signed char lb[2][64][128];

  const int tid   = threadIdx.x;
  const int lane  = tid & 31;
  const int wave  = tid >> 5;
  const int laneM = lane & 15;
  const int half  = lane >> 4;
  const int base  = half * 8;
  const int mbase = (blockIdx.y * 8 + wave) * 16;
  const int nbase = blockIdx.x * 64;

  const signed char* ap = A + (size_t)(mbase + laneM) * K;

  const int srow = tid >> 2, sseg = tid & 3;
  const unsigned long long wbase =
      (unsigned long long)(uintptr_t)(W + (size_t)(nbase + srow) * K + sseg * 32);
  const unsigned ldsb[2] = { lds_off(&lb[0][srow][sseg * 32]),
                             lds_off(&lb[1][srow][sseg * 32]) };

  v8i acc[4];
#pragma unroll
  for (int t = 0; t < 4; ++t)
#pragma unroll
    for (int i = 0; i < 8; ++i) acc[t][i] = 0;

  const int nc = K >> 7;
  async_copy32(ldsb[0], wbase);
  async_wait0();
  __syncthreads();

  for (int c = 0; c < nc; ++c) {
    const int k0 = c << 7;
    if (c + 1 < nc)
      async_copy32(ldsb[(c + 1) & 1], wbase + (unsigned long long)(k0 + 128));
    __builtin_prefetch(ap + k0 + 128, 0, 1);
    const signed char (*lbc)[128] = lb[c & 1];

#pragma unroll
    for (int kk = 0; kk < 128; kk += 64) {
      int2 a0 = *(const int2*)(ap + k0 + kk + base);
      int2 a1 = *(const int2*)(ap + k0 + kk + 16 + base);
      int2 a2 = *(const int2*)(ap + k0 + kk + 32 + base);
      int2 a3 = *(const int2*)(ap + k0 + kk + 48 + base);
      v8i a = {a0.x, a0.y, a1.x, a1.y, a2.x, a2.y, a3.x, a3.y};
      v8i b[4];
#pragma unroll
      for (int t = 0; t < 4; ++t) {
        const signed char* br = &lbc[t * 16 + laneM][kk];
        int2 b0 = *(const int2*)(br + base);
        int2 b1 = *(const int2*)(br + 16 + base);
        int2 b2 = *(const int2*)(br + 32 + base);
        int2 b3 = *(const int2*)(br + 48 + base);
        b[t] = (v8i){b0.x, b0.y, b1.x, b1.y, b2.x, b2.y, b3.x, b3.y};
      }
#pragma unroll
      for (int t = 0; t < 4; ++t)
        acc[t] = __builtin_amdgcn_wmma_i32_16x16x64_iu8(true, a, true, b[t],
                                                        acc[t], false, false);
    }
    async_wait0();
    __syncthreads();
  }

  const float sw = *swp;
  float inv[8];
#pragma unroll
  for (int i = 0; i < 8; ++i) inv[i] = 1.0f / (sx[mbase + i + 8 * half] * sw);

#pragma unroll
  for (int t = 0; t < 4; ++t) {
    const int col = nbase + t * 16 + laneM;
#pragma unroll
    for (int i = 0; i < 8; ++i) {
      const int row = mbase + i + 8 * half;
      float v = (float)acc[t][i] * inv[i];
      if (RESID) v += resid[(size_t)row * N + col];
      if (OUTF16) ((_Float16*)out)[(size_t)row * N + col] = (_Float16)v;
      else        ((float*)out)[(size_t)row * N + col] = v;
    }
  }
}

__global__ __launch_bounds__(256) void gemm_f16_kernel(
    const _Float16* __restrict__ A, const _Float16* __restrict__ W,
    _Float16* __restrict__ O, int M, int N, int K) {
  __shared__ _Float16 lbh[2][64][64];

  const int tid   = threadIdx.x;
  const int lane  = tid & 31;
  const int wave  = tid >> 5;
  const int laneM = lane & 15;
  const int half  = lane >> 4;
  const int base  = half * 8;
  const int mbase = (blockIdx.y * 8 + wave) * 16;
  const int nbase = blockIdx.x * 64;

  const _Float16* ap = A + (size_t)(mbase + laneM) * K;

  const int srow = tid >> 2, sseg = tid & 3;
  const unsigned long long wbase =
      (unsigned long long)(uintptr_t)(W + (size_t)(nbase + srow) * K + sseg * 16);
  const unsigned ldsb[2] = { lds_off(&lbh[0][srow][sseg * 16]),
                             lds_off(&lbh[1][srow][sseg * 16]) };

  v8f acc[4];
#pragma unroll
  for (int t = 0; t < 4; ++t)
#pragma unroll
    for (int i = 0; i < 8; ++i) acc[t][i] = 0.f;

  const int nc = K >> 6;
  async_copy32(ldsb[0], wbase);
  async_wait0();
  __syncthreads();

  for (int c = 0; c < nc; ++c) {
    const int k0 = c << 6;
    if (c + 1 < nc)
      async_copy32(ldsb[(c + 1) & 1], wbase + (unsigned long long)(k0 + 64) * 2);
    __builtin_prefetch(ap + k0 + 64, 0, 1);
    const _Float16 (*lbc)[64] = lbh[c & 1];

#pragma unroll
    for (int kk = 0; kk < 64; kk += 32) {
      v8h alo = *(const v8h*)(ap + k0 + kk + base);
      v8h ahi = *(const v8h*)(ap + k0 + kk + 16 + base);
      v16h a = cat8(alo, ahi);
      v16h b[4];
#pragma unroll
      for (int t = 0; t < 4; ++t) {
        const _Float16* br = &lbc[t * 16 + laneM][kk];
        b[t] = cat8(*(const v8h*)(br + base), *(const v8h*)(br + 16 + base));
      }
#pragma unroll
      for (int t = 0; t < 4; ++t)
        acc[t] = __builtin_amdgcn_wmma_f32_16x16x32_f16(false, a, false, b[t],
                                                        (short)0, acc[t], false, false);
    }
    async_wait0();
    __syncthreads();
  }
#pragma unroll
  for (int t = 0; t < 4; ++t) {
    const int col = nbase + t * 16 + laneM;
#pragma unroll
    for (int i = 0; i < 8; ++i) {
      const int row = mbase + i + 8 * half;
      O[(size_t)row * N + col] = (_Float16)acc[t][i];
    }
  }
}

__global__ __launch_bounds__(256) void rope_kernel(
    const _Float16* __restrict__ qkv, _Float16* __restrict__ Qo,
    _Float16* __restrict__ Ko, _Float16* __restrict__ Vo) {
  const int idx = blockIdx.x * 256 + threadIdx.x;
  const int d = idx & 63;
  const int h = (idx >> 6) & 15;
  const int s = (idx >> 10) & 4095;
  const int b = idx >> 22;
  const float inv = __expf(-(float)d * (9.210340371976184f / 64.0f));
  float sn, cs;
  __sincosf((float)s * inv, &sn, &cs);

  const _Float16* src = qkv + ((size_t)b * S_ + s) * QKV_N;
  const size_t ob = (((size_t)b * NH_ + h) * S_ + s) * HD_;
  float q1 = (float)src[0 * H_ + h * HD_ + d];
  float q2 = (float)src[0 * H_ + h * HD_ + d + 64];
  float k1 = (float)src[1 * H_ + h * HD_ + d];
  float k2 = (float)src[1 * H_ + h * HD_ + d + 64];
  Qo[ob + d]      = (_Float16)(q1 * cs - q2 * sn);
  Qo[ob + d + 64] = (_Float16)(q2 * cs + q1 * sn);
  Ko[ob + d]      = (_Float16)(k1 * cs - k2 * sn);
  Ko[ob + d + 64] = (_Float16)(k2 * cs + k1 * sn);
  Vo[ob + d]      = src[2 * H_ + h * HD_ + d];
  Vo[ob + d + 64] = src[2 * H_ + h * HD_ + d + 64];
}

__global__ __launch_bounds__(256) void attn_kernel(
    const _Float16* __restrict__ Q, const _Float16* __restrict__ K16,
    const _Float16* __restrict__ V16, float* __restrict__ ctx) {
  __shared__ _Float16 lk[32][128];
  __shared__ _Float16 lv[128][32];
  __shared__ _Float16 lp[8][16][32];

  const int tid   = threadIdx.x;
  const int lane  = tid & 31;
  const int wave  = tid >> 5;
  const int laneM = lane & 15;
  const int half  = lane >> 4;
  const int base  = half * 8;

  const int qc  = blockIdx.x;
  const int blk = blockIdx.y;
  const int bh  = blockIdx.z;

  const int qi0   = qc * 128 + wave * 16;
  const int cmax  = (qi0 + 15) >> 5;
  const int qrow0 = blk * BT_ + qi0;

  const _Float16* qp = Q + ((size_t)bh * S_ + qrow0 + laneM) * HD_;
  v16h qf[4];
#pragma unroll
  for (int ks = 0; ks < 4; ++ks) {
    v8h lo = *(const v8h*)(qp + ks * 32 + base);
    v8h hi = *(const v8h*)(qp + ks * 32 + 16 + base);
    qf[ks] = cat8(lo, hi);
  }

  const _Float16* kp = K16 + ((size_t)bh * S_ + blk * BT_) * HD_;
  const _Float16* vp = V16 + ((size_t)bh * S_ + blk * BT_) * HD_;

  v8f o[8], mrow, lrow;
#pragma unroll
  for (int d = 0; d < 8; ++d)
#pragma unroll
    for (int i = 0; i < 8; ++i) o[d][i] = 0.f;
#pragma unroll
  for (int i = 0; i < 8; ++i) { mrow[i] = -3.0e38f; lrow[i] = 0.f; }

  const int kvr = tid >> 3;
  const int seg = tid & 7;

  for (int c = 0; c < 16; ++c) {
    __syncthreads();
    v16h kk = *(const v16h*)(kp + (size_t)(c * 32 + kvr) * HD_ + seg * 16);
    *(v16h*)&lk[kvr][seg * 16] = kk;
    v16h vv = *(const v16h*)(vp + (size_t)(c * 32 + kvr) * HD_ + seg * 16);
#pragma unroll
    for (int j = 0; j < 16; ++j) lv[seg * 16 + j][kvr] = vv[j];
    __syncthreads();

    if (c <= cmax) {
      v8f st[2];
#pragma unroll
      for (int t2 = 0; t2 < 2; ++t2) {
        v8f s;
#pragma unroll
        for (int i = 0; i < 8; ++i) s[i] = 0.f;
#pragma unroll
        for (int ks = 0; ks < 4; ++ks) {
          v8h klo = *(const v8h*)&lk[t2 * 16 + laneM][ks * 32 + base];
          v8h khi = *(const v8h*)&lk[t2 * 16 + laneM][ks * 32 + 16 + base];
          s = __builtin_amdgcn_wmma_f32_16x16x32_f16(false, qf[ks], false,
                                                     cat8(klo, khi), (short)0,
                                                     s, false, false);
        }
        const int kj = c * 32 + t2 * 16 + laneM;
#pragma unroll
        for (int i = 0; i < 8; ++i) {
          s[i] *= 0.08838834764831845f;
          if (kj > qi0 + i + 8 * half) s[i] = -3.0e38f;
        }
        st[t2] = s;
      }
      v8f mx;
#pragma unroll
      for (int i = 0; i < 8; ++i) mx[i] = fmaxf(st[0][i], st[1][i]);
#pragma unroll
      for (int msk = 1; msk <= 8; msk <<= 1)
#pragma unroll
        for (int i = 0; i < 8; ++i) mx[i] = fmaxf(mx[i], __shfl_xor(mx[i], msk, 32));
      v8f mnew, alpha;
#pragma unroll
      for (int i = 0; i < 8; ++i) {
        mnew[i]  = fmaxf(mrow[i], mx[i]);
        alpha[i] = __expf(mrow[i] - mnew[i]);
      }
#pragma unroll
      for (int t2 = 0; t2 < 2; ++t2)
#pragma unroll
        for (int i = 0; i < 8; ++i) st[t2][i] = __expf(st[t2][i] - mnew[i]);
      v8f rs;
#pragma unroll
      for (int i = 0; i < 8; ++i) rs[i] = st[0][i] + st[1][i];
#pragma unroll
      for (int msk = 1; msk <= 8; msk <<= 1)
#pragma unroll
        for (int i = 0; i < 8; ++i) rs[i] += __shfl_xor(rs[i], msk, 32);
#pragma unroll
      for (int i = 0; i < 8; ++i) {
        lrow[i] = lrow[i] * alpha[i] + rs[i];
        mrow[i] = mnew[i];
      }
#pragma unroll
      for (int d = 0; d < 8; ++d)
#pragma unroll
        for (int i = 0; i < 8; ++i) o[d][i] *= alpha[i];

#pragma unroll
      for (int t2 = 0; t2 < 2; ++t2)
#pragma unroll
        for (int i = 0; i < 8; ++i)
          lp[wave][i + 8 * half][t2 * 16 + laneM] = (_Float16)st[t2][i];
      v8h plo = *(const v8h*)&lp[wave][laneM][base];
      v8h phi = *(const v8h*)&lp[wave][laneM][16 + base];
      v16h pf = cat8(plo, phi);

#pragma unroll
      for (int d = 0; d < 8; ++d) {
        v8h vlo = *(const v8h*)&lv[d * 16 + laneM][base];
        v8h vhi = *(const v8h*)&lv[d * 16 + laneM][16 + base];
        o[d] = __builtin_amdgcn_wmma_f32_16x16x32_f16(false, pf, false,
                                                      cat8(vlo, vhi), (short)0,
                                                      o[d], false, false);
      }
    }
  }

  const int b = bh >> 4, h = bh & 15;
#pragma unroll
  for (int d = 0; d < 8; ++d)
#pragma unroll
    for (int i = 0; i < 8; ++i) {
      const int srow = qrow0 + i + 8 * half;
      ctx[((size_t)b * S_ + srow) * H_ + h * HD_ + d * 16 + laneM] =
          o[d][i] / lrow[i];
    }
}

extern "C" void kernel_launch(void* const* d_in, const int* in_sizes, int n_in,
                              void* d_out, int out_size, void* d_ws, size_t ws_size,
                              hipStream_t stream) {
  (void)in_sizes; (void)n_in; (void)out_size; (void)ws_size;
  const float* x       = (const float*)d_in[0];
  const float* anorm   = (const float*)d_in[1];
  const float* fnorm   = (const float*)d_in[2];
  const float* qkv_w   = (const float*)d_in[3];
  const float* o_w     = (const float*)d_in[4];
  const float* up_w    = (const float*)d_in[5];
  const float* down_w  = (const float*)d_in[6];
  float* out = (float*)d_out;

  char* w = (char*)d_ws;
  size_t off = 0;
  auto alloc = [&](size_t bytes) { char* p = w + off; off = (off + bytes + 255) & ~(size_t)255; return p; };

  float*       scales = (float*)      alloc(8 * 4);
  float*       part   = (float*)      alloc(1024 * 4);
  signed char* wq_qkv = (signed char*)alloc((size_t)QKV_N * H_);
  signed char* wq_o   = (signed char*)alloc((size_t)H_ * H_);
  signed char* wq_dn  = (signed char*)alloc((size_t)H_ * INTER_);
  _Float16*    wup16  = (_Float16*)   alloc((size_t)UP_N * H_ * 2);
  signed char* xq     = (signed char*)alloc((size_t)M_ * H_);
  float*       sx     = (float*)      alloc((size_t)M_ * 4);
  _Float16*    qkv16  = (_Float16*)   alloc((size_t)M_ * QKV_N * 2);
  _Float16*    q16    = (_Float16*)   alloc((size_t)M_ * H_ * 2);
  _Float16*    k16    = (_Float16*)   alloc((size_t)M_ * H_ * 2);
  _Float16*    v16    = (_Float16*)   alloc((size_t)M_ * H_ * 2);
  float*       ctx    = (float*)      alloc((size_t)M_ * H_ * 4);
  signed char* ctxq   = (signed char*)alloc((size_t)M_ * H_);
  float*       sctx   = (float*)      alloc((size_t)M_ * 4);
  _Float16*    hn16   = (_Float16*)   alloc((size_t)M_ * H_ * 2);
  _Float16*    gv16   = (_Float16*)   alloc((size_t)M_ * UP_N * 2);
  signed char* tq     = (signed char*)alloc((size_t)M_ * INTER_);
  float*       st     = (float*)      alloc((size_t)M_ * 4);

  {
    const int n = QKV_N * H_;
    absum_part_kernel<<<192, 256, 0, stream>>>(qkv_w, part, n);
    absum_final_kernel<<<1, 256, 0, stream>>>(part, 192, &scales[0], (float)n);
    wquant_kernel<<<2048, 256, 0, stream>>>(qkv_w, &scales[0], wq_qkv, n);
  }
  {
    const int n = H_ * H_;
    absum_part_kernel<<<64, 256, 0, stream>>>(o_w, part, n);
    absum_final_kernel<<<1, 256, 0, stream>>>(part, 64, &scales[1], (float)n);
    wquant_kernel<<<2048, 256, 0, stream>>>(o_w, &scales[1], wq_o, n);
  }
  {
    const int n = H_ * INTER_;
    absum_part_kernel<<<128, 256, 0, stream>>>(down_w, part, n);
    absum_final_kernel<<<1, 256, 0, stream>>>(part, 128, &scales[2], (float)n);
    wquant_kernel<<<2048, 256, 0, stream>>>(down_w, &scales[2], wq_dn, n);
  }
  cvt16_kernel<<<2048, 256, 0, stream>>>(up_w, wup16, UP_N * H_);

  rowop_kernel<0><<<M_, 256, 0, stream>>>(x, anorm, xq, nullptr, sx);

  gemm_iu8_kernel<true, false><<<dim3(QKV_N / 64, M_ / 128), 256, 0, stream>>>(
      xq, wq_qkv, sx, &scales[0], nullptr, qkv16, M_, QKV_N, H_);

  rope_kernel<<<(B_ * S_ * NH_ * 64) / 256, 256, 0, stream>>>(qkv16, q16, k16, v16);

  attn_kernel<<<dim3(4, NB_, B_ * NH_), 256, 0, stream>>>(q16, k16, v16, ctx);

  rowop_kernel<1><<<M_, 256, 0, stream>>>(ctx, nullptr, ctxq, nullptr, sctx);

  gemm_iu8_kernel<false, true><<<dim3(H_ / 64, M_ / 128), 256, 0, stream>>>(
      ctxq, wq_o, sctx, &scales[1], x, out, M_, H_, H_);

  rowop_kernel<2><<<M_, 256, 0, stream>>>(out, fnorm, nullptr, hn16, nullptr);

  gemm_f16_kernel<<<dim3(UP_N / 64, M_ / 128), 256, 0, stream>>>(
      hn16, wup16, gv16, M_, UP_N, H_);

  silu_quant_kernel<<<M_, 256, 0, stream>>>(gv16, tq, st);

  gemm_iu8_kernel<false, true><<<dim3(H_ / 64, M_ / 128), 256, 0, stream>>>(
      tq, wq_dn, st, &scales[2], out, out, M_, H_, INTER_);
}
